// MessagePassingNeuralNetwork_5523327942769
// MI455X (gfx1250) — matrix operand order repaired and checked by reading, not compiled, not run on hardware
//
#include <hip/hip_runtime.h>
#include <hip/hip_bf16.h>

// ---------------- problem constants (match reference) ----------------
#define NN      50000      // nodes
#define NE      800000     // edges
#define TPASS   3
#define ED      64
#define NSH     23
#define MDIM    87         // NSH + ED
#define HID     200
#define NMOL    2048
#define RDIM    256        // ED*(T+1)
#define EPSBN   1e-5f

// padded dims for WMMA (K mult of 32, N mult of 16)
#define KP_MSG  96         // 87 -> 96
#define KP_HID  224        // 200 -> 224 (K side)
#define NP_HID  208        // 200 -> 208 (N side)
#define LDC_H   224        // stride of hidden activation buffers
#define MTILES  (NN/16)    // 3125 exactly

typedef __attribute__((ext_vector_type(16))) _Float16 v16h;
typedef __attribute__((ext_vector_type(8)))  _Float16 v8h;
typedef __attribute__((ext_vector_type(8)))  float    v8f;

// ---------------- utility kernels ----------------
__global__ void k_zero_f32(float* p, int n) {
    int i = blockIdx.x * 256 + threadIdx.x;
    if (i < n) p[i] = 0.f;
}
__global__ void k_zero_f16(_Float16* p, int n) {
    int i = blockIdx.x * 256 + threadIdx.x;
    if (i < n) p[i] = (_Float16)0.f;
}

// x_i = emb[z_i]  -> write into xall column block 0
__global__ void k_embed(const int* __restrict__ z, const float* __restrict__ emb,
                        float* __restrict__ xall) {
    int i = blockIdx.x * 256 + threadIdx.x;          // over NN*ED
    if (i >= NN * ED) return;
    int node = i / ED, c = i % ED;
    xall[(long long)node * RDIM + c] = emb[z[node] * ED + c];
}

// per-edge message scatter: m[src] += [rbf(dist), x[sink]]
__global__ void k_scatter(const int* __restrict__ esrc, const int* __restrict__ esnk,
                          const float* __restrict__ dist, const float* __restrict__ xall,
                          int xoff, float* __restrict__ m) {
    int i = blockIdx.x * 256 + threadIdx.x;          // over NE*MDIM (69.6M)
    if (i >= NE * MDIM) return;
    int e = i / MDIM, c = i % MDIM;
    float v;
    if (c < NSH) {
        float d = dist[e] - (0.8f + 0.1f * (float)c);
        v = expf(-d * d);
    } else {
        v = xall[(long long)esnk[e] * RDIM + xoff + (c - NSH)];
    }
    atomicAdd(&m[(long long)esrc[e] * MDIM + c], v);
}

// batch-norm statistics: sums[c], sums[C+c] accumulate column sum / sumsq
__global__ void k_bnstats(const float* __restrict__ X, int C, float* __restrict__ sums) {
    int col = blockIdx.y * 128 + threadIdx.x;
    if (col >= C) return;
    int r0 = blockIdx.x * 512;
    int r1 = r0 + 512; if (r1 > NN) r1 = NN;
    float s = 0.f, s2 = 0.f;
    for (int r = r0; r < r1; ++r) {
        float v = X[(long long)r * C + col];
        s += v; s2 += v * v;
    }
    atomicAdd(&sums[col], s);
    atomicAdd(&sums[C + col], s2);
}

__global__ void k_bnfin(const float* __restrict__ sums, const float* __restrict__ g,
                        const float* __restrict__ b, int C,
                        float* __restrict__ scale, float* __restrict__ shift) {
    int c = blockIdx.x * 256 + threadIdx.x;
    if (c >= C) return;
    float inv = 1.f / (float)NN;
    float mu  = sums[c] * inv;
    float var = sums[C + c] * inv - mu * mu;
    float rs  = rsqrtf(var + EPSBN);
    float sc  = rs * g[c];
    scale[c] = sc;
    shift[c] = b[c] - mu * sc;
}

// normalize fp32 -> padded fp16 A matrix (stride KP, pad cols zeroed)
__global__ void k_norm_f16(const float* __restrict__ X, int C, int ldx,
                           const float* __restrict__ scale, const float* __restrict__ shift,
                           _Float16* __restrict__ A, int KP) {
    long long i = (long long)blockIdx.x * 256 + threadIdx.x;   // over NN*KP
    if (i >= (long long)NN * KP) return;
    int r = (int)(i / KP), c = (int)(i % KP);
    float v = (c < C) ? X[(long long)r * ldx + c] * scale[c] + shift[c] : 0.f;
    A[i] = (_Float16)v;
}

// pad+convert fp32 weight [K x Nout] -> fp16 [KPp x NP]
__global__ void k_wpad(const float* __restrict__ W, int K, int Nout, int NP,
                       _Float16* __restrict__ Wp, int KPp) {
    int i = blockIdx.x * 256 + threadIdx.x;
    if (i >= KPp * NP) return;
    int k = i / NP, n = i % NP;
    Wp[i] = (k < K && n < Nout) ? (_Float16)W[k * Nout + n] : (_Float16)0.f;
}

// ---------------- shared WMMA inner core ----------------
// A: [NN x K] f16 row-major; W panel (64 cols) staged transposed in LDS.
// Each wave: 16x64 output strip = 4 accumulators; per K-step load one A
// fragment + 4 *distinct-register* B fragments, then 4 back-to-back WMMAs.
template <int K>
__device__ __forceinline__ void wmma_strip(const _Float16* __restrict__ Arow,
                                           const _Float16* __restrict__ wt,
                                           int laneRow, int laneHalf, v8f acc[4]) {
#pragma unroll
    for (int k0 = 0; k0 < K; k0 += 32) {
        if (k0 + 32 < K) __builtin_prefetch(Arow + k0 + 32, 0, 3);
        v8h alo = *(const v8h*)(Arow + k0 + laneHalf * 8);
        v8h ahi = *(const v8h*)(Arow + k0 + 16 + laneHalf * 8);
        v16h a;
#pragma unroll
        for (int j = 0; j < 8; ++j) { a[j] = alo[j]; a[j + 8] = ahi[j]; }

        v16h bfrag[4];
#pragma unroll
        for (int u = 0; u < 4; ++u) {
            const _Float16* wrow = &wt[(u * 16 + laneRow) * 256 + k0 + laneHalf * 8];
            v8h blo = *(const v8h*)(wrow);
            v8h bhi = *(const v8h*)(wrow + 16);
#pragma unroll
            for (int j = 0; j < 8; ++j) { bfrag[u][j] = blo[j]; bfrag[u][j + 8] = bhi[j]; }
        }
#pragma unroll
        for (int u = 0; u < 4; ++u) {
            acc[u] = __builtin_amdgcn_wmma_f32_16x16x32_f16(false, a, false, bfrag[u],
                                                            (short)0, acc[u], false, false);
        }
    }
}

// ---------------- WMMA GEMM: C = relu(A @ Wp + bias), f16 out ----------------
template <int K>
__global__ void __launch_bounds__(256)
k_gemm_relu(const _Float16* __restrict__ A,
            const _Float16* __restrict__ Wp, int NP,
            const float* __restrict__ bias, int Nout,
            _Float16* __restrict__ C, int ldc) {
    __shared__ _Float16 wt[64 * 256];   // transposed panel: wt[n*256 + k], 32 KB
    int tid = threadIdx.x;
    int ncol0 = blockIdx.y * 64;
#pragma unroll 4
    for (int idx = tid; idx < K * 64; idx += 256) {
        int k = idx >> 6, n = idx & 63;
        int col = ncol0 + n;
        wt[n * 256 + k] = (col < NP) ? Wp[k * NP + col] : (_Float16)0.f;
    }
    __syncthreads();

    int wave = tid >> 5, lane = tid & 31;
    int laneRow = lane & 15, laneHalf = lane >> 4;
    int tileM = blockIdx.x * 8 + wave;
    if (tileM >= MTILES) return;

    const _Float16* Arow = A + (long long)(tileM * 16 + laneRow) * K;
    v8f acc[4] = {};
    wmma_strip<K>(Arow, wt, laneRow, laneHalf, acc);

#pragma unroll
    for (int u = 0; u < 4; ++u) {
        int col = ncol0 + u * 16 + laneRow;
        if (col < Nout) {
            float bv = bias[col];
#pragma unroll
            for (int r = 0; r < 8; ++r) {
                int row = tileM * 16 + laneHalf * 8 + r;
                float v = acc[u][r] + bv;
                v = v > 0.f ? v : 0.f;
                C[(long long)row * ldc + col] = (_Float16)v;
            }
        }
    }
}

// ---------------- WMMA GEMM layer-4 + residual update ----------------
// delta = A @ W4p + b4 ; xall[:, (t+1)*64] = xall[:, t*64] + 0.1*delta
template <int K>
__global__ void __launch_bounds__(256)
k_gemm_update(const _Float16* __restrict__ A,
              const _Float16* __restrict__ Wp,   // [K x 64]
              const float* __restrict__ bias,
              float* __restrict__ xall, int tcur) {
    __shared__ _Float16 wt[64 * 256];
    int tid = threadIdx.x;
#pragma unroll 4
    for (int idx = tid; idx < K * 64; idx += 256) {
        int k = idx >> 6, n = idx & 63;
        wt[n * 256 + k] = Wp[k * 64 + n];
    }
    __syncthreads();

    int wave = tid >> 5, lane = tid & 31;
    int laneRow = lane & 15, laneHalf = lane >> 4;
    int tileM = blockIdx.x * 8 + wave;
    if (tileM >= MTILES) return;

    const _Float16* Arow = A + (long long)(tileM * 16 + laneRow) * K;
    v8f acc[4] = {};
    wmma_strip<K>(Arow, wt, laneRow, laneHalf, acc);

#pragma unroll
    for (int u = 0; u < 4; ++u) {
        int col = u * 16 + laneRow;          // 0..63
        float bv = bias[col];
#pragma unroll
        for (int r = 0; r < 8; ++r) {
            int row = tileM * 16 + laneHalf * 8 + r;
            long long base = (long long)row * RDIM;
            float xo = xall[base + tcur * ED + col];
            xall[base + (tcur + 1) * ED + col] = xo + 0.1f * (acc[u][r] + bv);
        }
    }
}

// readout last layer (Nout=1) + molecule segment-sum
__global__ void k_readout_final(const _Float16* __restrict__ h,
                                const float* __restrict__ w4, const float* __restrict__ b4,
                                const int* __restrict__ mol, float* __restrict__ out) {
    int i = blockIdx.x * 256 + threadIdx.x;
    if (i >= NN) return;
    const _Float16* hr = h + (long long)i * LDC_H;
    float acc = 0.f;
    for (int j = 0; j < HID; ++j) acc += (float)hr[j] * w4[j];
    acc += b4[0];
    atomicAdd(&out[mol[i]], acc);
}

// ---------------- host launcher ----------------
static inline dim3 g1(long long n) { return dim3((unsigned)((n + 255) / 256)); }

extern "C" void kernel_launch(void* const* d_in, const int* in_sizes, int n_in,
                              void* d_out, int out_size, void* d_ws, size_t ws_size,
                              hipStream_t stream) {
    const int*   z      = (const int*)d_in[0];
    const int*   eidx   = (const int*)d_in[1];
    const int*   esrc   = eidx;            // edge_index[0]
    const int*   esnk   = eidx + NE;       // edge_index[1]
    const float* dist   = (const float*)d_in[2];
    const int*   mol    = (const int*)d_in[3];
    const float* emb    = (const float*)d_in[4];
    const float* upg    = (const float*)d_in[5];
    const float* upb    = (const float*)d_in[6];
    const float* uw1    = (const float*)d_in[7];
    const float* ub1    = (const float*)d_in[8];
    const float* uw2    = (const float*)d_in[9];
    const float* ub2    = (const float*)d_in[10];
    const float* uw3    = (const float*)d_in[11];
    const float* ub3    = (const float*)d_in[12];
    const float* uw4    = (const float*)d_in[13];
    const float* ub4    = (const float*)d_in[14];
    const float* rog    = (const float*)d_in[15];
    const float* rob    = (const float*)d_in[16];
    const float* rw1    = (const float*)d_in[17];
    const float* rb1    = (const float*)d_in[18];
    const float* rw2    = (const float*)d_in[19];
    const float* rb2    = (const float*)d_in[20];
    const float* rw3    = (const float*)d_in[21];
    const float* rb3    = (const float*)d_in[22];
    const float* rw4    = (const float*)d_in[23];
    const float* rb4    = (const float*)d_in[24];
    float* out = (float*)d_out;

    // workspace carve-up
    char* ws = (char*)d_ws;
    size_t off = 0;
    auto carve = [&](size_t bytes) -> void* {
        void* p = ws + off;
        off = (off + bytes + 255) & ~(size_t)255;
        return p;
    };
    float*     m     = (float*)    carve((size_t)NN * MDIM * 4);
    _Float16*  Abuf  = (_Float16*) carve((size_t)NN * RDIM * 2);
    _Float16*  h0    = (_Float16*) carve((size_t)NN * LDC_H * 2);
    _Float16*  h1    = (_Float16*) carve((size_t)NN * LDC_H * 2);
    float*     xall  = (float*)    carve((size_t)NN * RDIM * 4);
    float*     sums  = (float*)    carve(512 * 4);   // sum + sumsq (<=256 cols)
    float*     scale = (float*)    carve(256 * 4);
    float*     shift = (float*)    carve(256 * 4);
    _Float16*  w1p   = (_Float16*) carve((size_t)KP_MSG * NP_HID * 2);
    _Float16*  w2p   = (_Float16*) carve((size_t)KP_HID * NP_HID * 2);
    _Float16*  w3p   = (_Float16*) carve((size_t)KP_HID * NP_HID * 2);
    _Float16*  w4p   = (_Float16*) carve((size_t)KP_HID * ED * 2);
    _Float16*  r1p   = (_Float16*) carve((size_t)RDIM  * NP_HID * 2);
    _Float16*  r2p   = (_Float16*) carve((size_t)KP_HID * NP_HID * 2);
    _Float16*  r3p   = (_Float16*) carve((size_t)KP_HID * NP_HID * 2);

    const dim3 blk(256);
    const dim3 gemmGridH((MTILES + 7) / 8, (NP_HID + 63) / 64);  // (391, 4)
    const dim3 gemmGridE((MTILES + 7) / 8, 1);                   // layer 4

    // init: output accum + hidden pads
    k_zero_f32<<<g1(NMOL), blk, 0, stream>>>(out, NMOL);
    k_zero_f16<<<g1((long long)NN * LDC_H), blk, 0, stream>>>(h0, NN * LDC_H);
    k_zero_f16<<<g1((long long)NN * LDC_H), blk, 0, stream>>>(h1, NN * LDC_H);

    // embedding -> xall block 0
    k_embed<<<g1((long long)NN * ED), blk, 0, stream>>>(z, emb, xall);

    // readout weights (pass-invariant): convert once
    k_wpad<<<g1((long long)RDIM * NP_HID),  blk, 0, stream>>>(rw1, RDIM, HID, NP_HID, r1p, RDIM);
    k_wpad<<<g1((long long)KP_HID * NP_HID),blk, 0, stream>>>(rw2, HID,  HID, NP_HID, r2p, KP_HID);
    k_wpad<<<g1((long long)KP_HID * NP_HID),blk, 0, stream>>>(rw3, HID,  HID, NP_HID, r3p, KP_HID);

    for (int t = 0; t < TPASS; ++t) {
        // message scatter-sum
        k_zero_f32<<<g1((long long)NN * MDIM), blk, 0, stream>>>(m, NN * MDIM);
        k_scatter<<<g1((long long)NE * MDIM), blk, 0, stream>>>(esrc, esnk, dist, xall, t * ED, m);

        // batchnorm over m (C=87)
        k_zero_f32<<<g1(512), blk, 0, stream>>>(sums, 512);
        k_bnstats<<<dim3((NN + 511) / 512, 1), dim3(128), 0, stream>>>(m, MDIM, sums);
        k_bnfin<<<g1(MDIM), blk, 0, stream>>>(sums, upg + t * MDIM, upb + t * MDIM, MDIM, scale, shift);
        k_norm_f16<<<g1((long long)NN * KP_MSG), blk, 0, stream>>>(m, MDIM, MDIM, scale, shift, Abuf, KP_MSG);

        // per-pass weights -> padded f16
        k_wpad<<<g1((long long)KP_MSG * NP_HID), blk, 0, stream>>>(uw1 + (size_t)t * MDIM * HID, MDIM, HID, NP_HID, w1p, KP_MSG);
        k_wpad<<<g1((long long)KP_HID * NP_HID), blk, 0, stream>>>(uw2 + (size_t)t * HID * HID, HID, HID, NP_HID, w2p, KP_HID);
        k_wpad<<<g1((long long)KP_HID * NP_HID), blk, 0, stream>>>(uw3 + (size_t)t * HID * HID, HID, HID, NP_HID, w3p, KP_HID);
        k_wpad<<<g1((long long)KP_HID * ED),     blk, 0, stream>>>(uw4 + (size_t)t * HID * ED,  HID, ED,  ED,     w4p, KP_HID);

        // MLP chain (WMMA)
        k_gemm_relu<KP_MSG><<<gemmGridH, blk, 0, stream>>>(Abuf, w1p, NP_HID, ub1 + t * HID, HID, h0, LDC_H);
        k_gemm_relu<KP_HID><<<gemmGridH, blk, 0, stream>>>(h0, w2p, NP_HID, ub2 + t * HID, HID, h1, LDC_H);
        k_gemm_relu<KP_HID><<<gemmGridH, blk, 0, stream>>>(h1, w3p, NP_HID, ub3 + t * HID, HID, h0, LDC_H);
        k_gemm_update<KP_HID><<<gemmGridE, blk, 0, stream>>>(h0, w4p, ub4 + t * ED, xall, t);
    }

    // readout: batchnorm over xall (C=256) -> MLP -> molecule segment sum
    k_zero_f32<<<g1(512), blk, 0, stream>>>(sums, 512);
    k_bnstats<<<dim3((NN + 511) / 512, 2), dim3(128), 0, stream>>>(xall, RDIM, sums);
    k_bnfin<<<g1(RDIM), blk, 0, stream>>>(sums, rog, rob, RDIM, scale, shift);
    k_norm_f16<<<g1((long long)NN * RDIM), blk, 0, stream>>>(xall, RDIM, RDIM, scale, shift, Abuf, RDIM);

    k_gemm_relu<RDIM><<<gemmGridH, blk, 0, stream>>>(Abuf, r1p, NP_HID, rb1, HID, h0, LDC_H);
    k_gemm_relu<KP_HID><<<gemmGridH, blk, 0, stream>>>(h0, r2p, NP_HID, rb2, HID, h1, LDC_H);
    k_gemm_relu<KP_HID><<<gemmGridH, blk, 0, stream>>>(h1, r3p, NP_HID, rb3, HID, h0, LDC_H);
    k_readout_final<<<g1(NN), blk, 0, stream>>>(h0, rw4, rb4, mol, out);
}
